// EncoderBlock_54494545051968
// MI455X (gfx1250) — matrix operand order repaired and checked by reading, not compiled, not run on hardware
//
#include <hip/hip_runtime.h>
#include <hip/hip_bf16.h>
#include <math.h>

#define D_MODEL 1024
#define HEADS   16
#define DK      64
#define D_FF    2048
#define BATCH   8
#define SEQ     1024
#define MTOK    (BATCH*SEQ)   /* 8192 token rows */

typedef __attribute__((ext_vector_type(16))) __bf16 v16bf;
typedef __attribute__((ext_vector_type(8)))  float  v8f;
typedef __attribute__((ext_vector_type(4)))  __bf16 bf4;

union BFrag { v16bf v; uint4 u[2]; };
union VPack { uint4 u; __bf16 h[8]; };

// ---------------------------------------------------------------------------
// fp32 -> bf16 conversion (vectorized, 4 elems/thread)
// ---------------------------------------------------------------------------
__global__ __launch_bounds__(256) void cvt_bf16(const float* __restrict__ in,
                                                __bf16* __restrict__ out, int n) {
  int i = (blockIdx.x * 256 + threadIdx.x) * 4;
  if (i < n) {
    float4 f = *reinterpret_cast<const float4*>(in + i);
    bf4 o = { (__bf16)f.x, (__bf16)f.y, (__bf16)f.z, (__bf16)f.w };
    *reinterpret_cast<bf4*>(out + i) = o;
  }
}

// ---------------------------------------------------------------------------
// Generic WMMA GEMM:  out[M,N] = A[M,K] * W[N,K]^T + bias  (+resid) (ReLU)
// Block tile 128x256 (8 waves, 2x4), wave tile 64x64 -> 16 accumulators,
// 16 WMMAs per K-step with 16 b128 loads (1.0 loads/WMMA, 4x frag reuse).
// A fragment: lane row = m0+lane%16, elem e -> k = half*16+e (contiguous 32B)
// B fragment: lane col = n0+lane%16, elem e -> k = half*16+e (contiguous 32B)
// ---------------------------------------------------------------------------
template<bool RELU, bool RESID, bool OUTBF>
__global__ __launch_bounds__(256) void gemm_wmma(
    const __bf16* __restrict__ A, const __bf16* __restrict__ W,
    const float* __restrict__ bias, const float* __restrict__ resid,
    float* __restrict__ outF, __bf16* __restrict__ outB, int N, int K) {
  const int lane = threadIdx.x & 31;
  const int wave = threadIdx.x >> 5;
  const int rl   = lane & 15;
  const int hf   = lane >> 4;
  const int wm   = wave & 1, wn = wave >> 1;
  const int m0   = blockIdx.x * 128 + wm * 64;
  const int n0   = blockIdx.y * 256 + wn * 64;

  const __bf16* pA[4];
  const __bf16* pB[4];
#pragma unroll
  for (int t = 0; t < 4; ++t) {
    pA[t] = A + (size_t)(m0 + t * 16 + rl) * K;
    pB[t] = W + (size_t)(n0 + t * 16 + rl) * K + hf * 16;
  }

  const v8f vzero = {0.f, 0.f, 0.f, 0.f, 0.f, 0.f, 0.f, 0.f};
  v8f acc[4][4];
#pragma unroll
  for (int mt = 0; mt < 4; ++mt)
#pragma unroll
    for (int nt = 0; nt < 4; ++nt) acc[mt][nt] = vzero;

  for (int k = 0; k < K; k += 32) {
    BFrag a[4], b[4];
#pragma unroll
    for (int t = 0; t < 4; ++t) {
      a[t].u[0] = *reinterpret_cast<const uint4*>(pA[t] + k + hf * 16);
      a[t].u[1] = *reinterpret_cast<const uint4*>(pA[t] + k + hf * 16 + 8);
      b[t].u[0] = *reinterpret_cast<const uint4*>(pB[t] + k);
      b[t].u[1] = *reinterpret_cast<const uint4*>(pB[t] + k + 8);
    }
    __builtin_prefetch(pA[0] + k + 256, 0, 0);   // global_prefetch_b8
    __builtin_prefetch(pB[0] + k + 256, 0, 0);
#pragma unroll
    for (int mt = 0; mt < 4; ++mt)
#pragma unroll
      for (int nt = 0; nt < 4; ++nt)
        acc[mt][nt] = __builtin_amdgcn_wmma_f32_16x16x32_bf16(
            false, a[mt].v, false, b[nt].v, (short)0, acc[mt][nt], false, false);
  }

  float bv[4];
#pragma unroll
  for (int t = 0; t < 4; ++t) bv[t] = bias[n0 + t * 16 + rl];

#pragma unroll
  for (int mt = 0; mt < 4; ++mt)
#pragma unroll
    for (int nt = 0; nt < 4; ++nt)
#pragma unroll
      for (int r = 0; r < 8; ++r) {
        const int row = m0 + mt * 16 + r + hf * 8;   // C layout: M = r + 8*half
        const int col = n0 + nt * 16 + rl;
        float v = acc[mt][nt][r] + bv[nt];
        if (RESID) v += resid[(size_t)row * N + col];
        if (RELU)  v = fmaxf(v, 0.f);
        if (OUTBF) outB[(size_t)row * N + col] = (__bf16)v;
        else       outF[(size_t)row * N + col] = v;
      }
}

// ---------------------------------------------------------------------------
// Flash attention: one block = (b, h, 128 queries); one wave = 16 queries.
// Online softmax, keys in chunks of 32. K chunk is staged into LDS with the
// CDNA5 async path (GLOBAL_LOAD_ASYNC_TO_LDS_B128 + s_wait_asynccnt); the V
// chunk goes through VGPRs because it is transposed on the way into LDS.
// ---------------------------------------------------------------------------
__global__ __launch_bounds__(256) void attn_flash(
    const __bf16* __restrict__ Q, const __bf16* __restrict__ Kg,
    const __bf16* __restrict__ Vg, const int* __restrict__ maskp,
    __bf16* __restrict__ O) {
  __shared__ __align__(16) __bf16 Kc[32][72];      // K chunk, row-major (144B stride)
  __shared__ __align__(16) __bf16 Vt[64][40];      // V chunk, transposed (80B stride)
  __shared__ __align__(16) __bf16 Pb[8][16][40];   // per-wave probability tiles

  const int lane = threadIdx.x & 31;
  const int wave = threadIdx.x >> 5;
  const int rl   = lane & 15, hf = lane >> 4;
  const int qb   = blockIdx.x & 7;
  const int bh   = blockIdx.x >> 3;
  const int b    = bh >> 4;
  const int h    = bh & 15;
  const int q0   = qb * 128 + wave * 16;

  // Q fragments for the two 32-wide dk steps
  const __bf16* pQ = Q + (size_t)(b * SEQ + q0 + rl) * D_MODEL + h * DK;
  BFrag qf[2];
#pragma unroll
  for (int d = 0; d < 2; ++d) {
    qf[d].u[0] = *reinterpret_cast<const uint4*>(pQ + d * 32 + hf * 16);
    qf[d].u[1] = *reinterpret_cast<const uint4*>(pQ + d * 32 + hf * 16 + 8);
  }

  const v8f vzero = {0.f, 0.f, 0.f, 0.f, 0.f, 0.f, 0.f, 0.f};
  v8f o[4];
#pragma unroll
  for (int j = 0; j < 4; ++j) o[j] = vzero;
  float mrow[8], lrow[8];
#pragma unroll
  for (int r = 0; r < 8; ++r) { mrow[r] = -1e30f; lrow[r] = 0.f; }

  const int tid  = threadIdx.x;
  const int srow = tid >> 3;          // 0..31
  const int scol = (tid & 7) * 8;     // 0..56
  // per-thread LDS byte offset for the async K-tile store (low 32 bits of the
  // generic LDS address are the within-group LDS byte address)
  const unsigned kc_lds = (unsigned)(unsigned long long)(void*)&Kc[srow][scol];

  for (int kc = 0; kc < SEQ / 32; ++kc) {
    const size_t g = (size_t)(b * SEQ + kc * 32 + srow) * D_MODEL + h * DK + scol;
    // K chunk: direct global -> LDS async copy (ASYNCcnt-tracked)
    {
      const __bf16* gp = Kg + g;
      asm volatile("global_load_async_to_lds_b128 %0, %1, off"
                   :: "v"(kc_lds), "v"(gp) : "memory");
    }
    // V chunk: load through VGPRs and store transposed
    {
      VPack vv; vv.u = *reinterpret_cast<const uint4*>(Vg + g);
#pragma unroll
      for (int j = 0; j < 8; ++j) Vt[scol + j][srow] = vv.h[j];
    }
    asm volatile("s_wait_asynccnt 0" ::: "memory");
    __syncthreads();

    // scores for two 16-key tiles: S = Q * Kc^T  (dk=64 -> two WMMA k-steps)
    v8f s[2];
#pragma unroll
    for (int nt = 0; nt < 2; ++nt) {
      v8f a = vzero;
      BFrag bk;
      bk.u[0] = *reinterpret_cast<const uint4*>(&Kc[nt * 16 + rl][hf * 16]);
      bk.u[1] = *reinterpret_cast<const uint4*>(&Kc[nt * 16 + rl][hf * 16 + 8]);
      a = __builtin_amdgcn_wmma_f32_16x16x32_bf16(false, qf[0].v, false, bk.v,
                                                  (short)0, a, false, false);
      bk.u[0] = *reinterpret_cast<const uint4*>(&Kc[nt * 16 + rl][32 + hf * 16]);
      bk.u[1] = *reinterpret_cast<const uint4*>(&Kc[nt * 16 + rl][32 + hf * 16 + 8]);
      a = __builtin_amdgcn_wmma_f32_16x16x32_bf16(false, qf[1].v, false, bk.v,
                                                  (short)0, a, false, false);
      s[nt] = a;
    }

    // scale + key mask (column-wise: this lane's key)
#pragma unroll
    for (int nt = 0; nt < 2; ++nt) {
      const int key = kc * 32 + nt * 16 + rl;
      const bool keep = maskp[b * SEQ + key] != 0;
#pragma unroll
      for (int e = 0; e < 8; ++e) {
        float v = s[nt][e] * 0.125f;           // 1/sqrt(64)
        s[nt][e] = keep ? v : -1.0e9f;
      }
    }

    // online softmax update (row = r + 8*half, spread over 16 lanes)
    float scl[8];
#pragma unroll
    for (int r = 0; r < 8; ++r) {
      float tmax = fmaxf(s[0][r], s[1][r]);
#pragma unroll
      for (int m = 1; m < 16; m <<= 1) tmax = fmaxf(tmax, __shfl_xor(tmax, m, 32));
      const float nm = fmaxf(mrow[r], tmax);
      scl[r] = __expf(mrow[r] - nm);
      const float p0 = __expf(s[0][r] - nm);
      const float p1 = __expf(s[1][r] - nm);
      s[0][r] = p0; s[1][r] = p1;
      float rs = p0 + p1;
#pragma unroll
      for (int m = 1; m < 16; m <<= 1) rs += __shfl_xor(rs, m, 32);
      lrow[r] = lrow[r] * scl[r] + rs;
      mrow[r] = nm;
    }
#pragma unroll
    for (int j = 0; j < 4; ++j)
#pragma unroll
      for (int r = 0; r < 8; ++r) o[j][r] *= scl[r];

    // spill probabilities (C layout) to LDS, reload in A layout
#pragma unroll
    for (int nt = 0; nt < 2; ++nt)
#pragma unroll
      for (int r = 0; r < 8; ++r)
        Pb[wave][r + hf * 8][nt * 16 + rl] = (__bf16)s[nt][r];

    BFrag ap;
    ap.u[0] = *reinterpret_cast<const uint4*>(&Pb[wave][rl][hf * 16]);
    ap.u[1] = *reinterpret_cast<const uint4*>(&Pb[wave][rl][hf * 16 + 8]);
#pragma unroll
    for (int j = 0; j < 4; ++j) {
      BFrag bv;
      bv.u[0] = *reinterpret_cast<const uint4*>(&Vt[j * 16 + rl][hf * 16]);
      bv.u[1] = *reinterpret_cast<const uint4*>(&Vt[j * 16 + rl][hf * 16 + 8]);
      o[j] = __builtin_amdgcn_wmma_f32_16x16x32_bf16(false, ap.v, false, bv.v,
                                                     (short)0, o[j], false, false);
    }
    __syncthreads();
  }

  // normalize and store O (bf16, [token][h*64+dim])
#pragma unroll
  for (int r = 0; r < 8; ++r) lrow[r] = 1.0f / lrow[r];
#pragma unroll
  for (int j = 0; j < 4; ++j)
#pragma unroll
    for (int r = 0; r < 8; ++r) {
      const int row = b * SEQ + q0 + r + hf * 8;
      const int col = h * DK + j * 16 + rl;
      O[(size_t)row * D_MODEL + col] = (__bf16)(o[j][r] * lrow[r]);
    }
}

// ---------------------------------------------------------------------------
// Reference's layer "norm": out = (z - (z-mean)/(std+eps))*alpha + beta, ddof=1
// One block per row of D_MODEL.
// ---------------------------------------------------------------------------
__global__ __launch_bounds__(256) void ln_kernel(
    const float* __restrict__ y, const float* __restrict__ alpha,
    const float* __restrict__ beta, float* __restrict__ out,
    __bf16* __restrict__ outb, int writeb) {
  const int D = D_MODEL;
  const float* row = y + (size_t)blockIdx.x * D;
  float v[4], s = 0.f, s2 = 0.f;
#pragma unroll
  for (int j = 0; j < 4; ++j) {
    v[j] = row[threadIdx.x + j * 256];
    s += v[j]; s2 += v[j] * v[j];
  }
#pragma unroll
  for (int m = 1; m < 32; m <<= 1) {
    s  += __shfl_xor(s, m, 32);
    s2 += __shfl_xor(s2, m, 32);
  }
  __shared__ float ws[8], ws2[8];
  if ((threadIdx.x & 31) == 0) { ws[threadIdx.x >> 5] = s; ws2[threadIdx.x >> 5] = s2; }
  __syncthreads();
  s = 0.f; s2 = 0.f;
#pragma unroll
  for (int w = 0; w < 8; ++w) { s += ws[w]; s2 += ws2[w]; }
  const float mean = s / (float)D;
  const float var = (s2 - (float)D * mean * mean) / (float)(D - 1);
  const float denom = sqrtf(fmaxf(var, 0.f)) + 1e-6f;
  float* orow = out + (size_t)blockIdx.x * D;
#pragma unroll
  for (int j = 0; j < 4; ++j) {
    const int c = threadIdx.x + j * 256;
    const float z = v[j];
    const float oo = (z - (z - mean) / denom) * alpha[c] + beta[c];
    orow[c] = oo;
    if (writeb) outb[(size_t)blockIdx.x * D + c] = (__bf16)oo;
  }
}

// ---------------------------------------------------------------------------
extern "C" void kernel_launch(void* const* d_in, const int* in_sizes, int n_in,
                              void* d_out, int out_size, void* d_ws, size_t ws_size,
                              hipStream_t stream) {
  (void)in_sizes; (void)n_in; (void)out_size; (void)ws_size;
  const float* x      = (const float*)d_in[0];
  const int*   maskp  = (const int*)d_in[1];
  const float* Wq = (const float*)d_in[2];  const float* bq = (const float*)d_in[3];
  const float* Wk = (const float*)d_in[4];  const float* bk = (const float*)d_in[5];
  const float* Wv = (const float*)d_in[6];  const float* bv = (const float*)d_in[7];
  const float* Wo = (const float*)d_in[8];  const float* bo = (const float*)d_in[9];
  const float* W1 = (const float*)d_in[10]; const float* b1 = (const float*)d_in[11];
  const float* W2 = (const float*)d_in[12]; const float* b2 = (const float*)d_in[13];
  const float* alpha1 = (const float*)d_in[14]; const float* beta1 = (const float*)d_in[15];
  const float* alpha2 = (const float*)d_in[16]; const float* beta2 = (const float*)d_in[17];

  char* w = (char*)d_ws;
  const size_t MB = 1ull << 20;
  __bf16* Wqb = (__bf16*)(w + 0 * MB);
  __bf16* Wkb = (__bf16*)(w + 2 * MB);
  __bf16* Wvb = (__bf16*)(w + 4 * MB);
  __bf16* Wob = (__bf16*)(w + 6 * MB);
  __bf16* W1b = (__bf16*)(w + 8 * MB);
  __bf16* W2b = (__bf16*)(w + 12 * MB);
  __bf16* xb  = (__bf16*)(w + 16 * MB);   // 16 MB; reused later as x1 bf16
  __bf16* x1b = xb;
  __bf16* Qb  = (__bf16*)(w + 32 * MB);
  __bf16* Kb  = (__bf16*)(w + 48 * MB);
  __bf16* Vb  = (__bf16*)(w + 64 * MB);
  __bf16* Hb  = (__bf16*)(w + 32 * MB);   // reuses Q/K/V region after attention
  __bf16* Ob  = (__bf16*)(w + 80 * MB);
  float*  y1  = (float*)(w + 96 * MB);    // 32 MB; reused as y2
  float*  y2  = y1;
  float*  x1  = (float*)(w + 128 * MB);   // 32 MB  (total ws use: 160 MB)

  const dim3 blk(256);
  auto cvt = [&](const float* src, __bf16* dst, int n) {
    cvt_bf16<<<dim3((n / 4 + 255) / 256), blk, 0, stream>>>(src, dst, n);
  };
  cvt(x,  xb,  MTOK * D_MODEL);
  cvt(Wq, Wqb, D_MODEL * D_MODEL);
  cvt(Wk, Wkb, D_MODEL * D_MODEL);
  cvt(Wv, Wvb, D_MODEL * D_MODEL);
  cvt(Wo, Wob, D_MODEL * D_MODEL);
  cvt(W1, W1b, D_FF * D_MODEL);
  cvt(W2, W2b, D_MODEL * D_FF);

  const dim3 gD(MTOK / 128, D_MODEL / 256);   // 64 x 4
  gemm_wmma<false, false, true><<<gD, blk, 0, stream>>>(xb, Wqb, bq, nullptr, nullptr, Qb, D_MODEL, D_MODEL);
  gemm_wmma<false, false, true><<<gD, blk, 0, stream>>>(xb, Wkb, bk, nullptr, nullptr, Kb, D_MODEL, D_MODEL);
  gemm_wmma<false, false, true><<<gD, blk, 0, stream>>>(xb, Wvb, bv, nullptr, nullptr, Vb, D_MODEL, D_MODEL);

  attn_flash<<<dim3(BATCH * HEADS * (SEQ / 128)), blk, 0, stream>>>(Qb, Kb, Vb, maskp, Ob);

  gemm_wmma<false, true, false><<<gD, blk, 0, stream>>>(Ob, Wob, bo, x, y1, nullptr, D_MODEL, D_MODEL);
  ln_kernel<<<dim3(MTOK), blk, 0, stream>>>(y1, alpha1, beta1, x1, x1b, 1);

  const dim3 gF(MTOK / 128, D_FF / 256);      // 64 x 8
  gemm_wmma<true, false, true><<<gF, blk, 0, stream>>>(x1b, W1b, b1, nullptr, nullptr, Hb, D_FF, D_MODEL);
  gemm_wmma<false, true, false><<<gD, blk, 0, stream>>>(Hb, W2b, b2, x1, y2, nullptr, D_MODEL, D_FF);

  ln_kernel<<<dim3(MTOK), blk, 0, stream>>>(y2, alpha2, beta2, (float*)d_out, nullptr, 0);
}
